// AtlasTransformerV5_64355789963703
// MI455X (gfx1250) — compile-verified, matrix operand order checked by reading, not run on hardware
//
#include <hip/hip_runtime.h>
#include <math.h>

#define B_    8
#define N_    512
#define H_    512
#define NH_   8
#define HD_   64
#define FF_   1024
#define NCH_  16
#define NPOS_ 32
#define NEDGE_ 16

typedef __attribute__((ext_vector_type(16))) _Float16 v16h;
typedef __attribute__((ext_vector_type(8)))  float    v8f;

#define NEG_INF (-__builtin_inff())

__device__ __forceinline__ float eluf(float x) { return x > 0.f ? x : __expf(x) - 1.f; }

// ============================================================================
// Generic strided batched WMMA GEMM:
//   z in grid.z, zo = z/zdiv, zi = z%zdiv
//   A base += zo*a0 + zi*a1 ; element (m,k)  = A[m*ldaM + k]
//   B base += zo*b0 + zi*b1 ; element (k,n)  = B[k*ldbK + n*ldbN]   (lets us do K^T)
//   C base += zo*c0 + zi*c1 ; element (m,n)  = C[m*ldc + n]
//   C = [addc? C +] act( scale * A@B + bias[n] )      act: 0=none 1=elu 2=tanh
// One wave -> one 16x16 C tile, f16 multiply / f32 accumulate via
// v_wmma_f32_16x16x32_f16.
// TAIL=false: K % 32 == 0 -> fully unconditional, vectorizable loads.
// TAIL=true : K may be < a multiple of 32 (only K=16 here) -> branch-free
//             clamped loads + select-to-zero (no exec-mask serialization).
// ============================================================================
template <bool TAIL>
__global__ void __launch_bounds__(128)
gemm_wmma(const float* __restrict__ A, long long a0, long long a1, int ldaM,
          const float* __restrict__ Bm, long long b0, long long b1, int ldbK, int ldbN,
          const float* __restrict__ bias,
          float* __restrict__ C, long long c0, long long c1, int ldc,
          int M, int N, int K, int zdiv, float scale, int act, int addc)
{
  int z  = blockIdx.z;
  int zo = z / zdiv, zi = z - zo * zdiv;
  const float* Ab = A  + (long long)zo * a0 + (long long)zi * a1;
  const float* Bb = Bm + (long long)zo * b0 + (long long)zi * b1;
  float*       Cb = C  + (long long)zo * c0 + (long long)zi * c1;

  int lane = threadIdx.x & 31;
  int wave = threadIdx.x >> 5;
  int tn = (blockIdx.x * 4 + wave) * 16;
  int tm = blockIdx.y * 16;
  if (tn >= N) return;            // uniform per wave -> EXEC stays all-ones for WMMA

  int l15  = lane & 15;
  int hi   = lane >> 4;           // 0 or 1
  int khi  = hi * 8;
  int arow = tm + l15;
  int bcol = tn + l15;

  const float* arowp = Ab + (long long)arow * ldaM;
  const float* bcolp = Bb + (long long)bcol * ldbN;

  v8f c = {0.f, 0.f, 0.f, 0.f, 0.f, 0.f, 0.f, 0.f};
  for (int k0 = 0; k0 < K; k0 += 32) {
    v16h a, b;
    if constexpr (!TAIL) {
      const float* ap = arowp + k0 + khi;
#pragma unroll
      for (int j = 0; j < 8; ++j) {
        a[j]     = (_Float16)ap[j];        // K = k0+khi+j       (contiguous -> b128)
        a[j + 8] = (_Float16)ap[j + 16];   // K = k0+khi+j+16    (contiguous -> b128)
        b[j]     = (_Float16)bcolp[(long long)(k0 + khi + j) * ldbK];
        b[j + 8] = (_Float16)bcolp[(long long)(k0 + khi + j + 16) * ldbK];
      }
    } else {
#pragma unroll
      for (int j = 0; j < 8; ++j) {
        int k1 = k0 + khi + j;
        int k2 = k1 + 16;
        int k1c = k1 < K ? k1 : K - 1;     // clamp: load stays unconditional
        int k2c = k2 < K ? k2 : K - 1;
        float av1 = arowp[k1c];
        float av2 = arowp[k2c];
        float bv1 = bcolp[(long long)k1c * ldbK];
        float bv2 = bcolp[(long long)k2c * ldbK];
        a[j]     = (_Float16)(k1 < K ? av1 : 0.f);
        a[j + 8] = (_Float16)(k2 < K ? av2 : 0.f);
        b[j]     = (_Float16)(k1 < K ? bv1 : 0.f);
        b[j + 8] = (_Float16)(k2 < K ? bv2 : 0.f);
      }
    }
    c = __builtin_amdgcn_wmma_f32_16x16x32_f16(false, a, false, b, (short)0, c, false, false);
  }

  int rbase = tm + hi * 8;        // C layout: vgpr r -> row r (+8 for lanes 16..31)
#pragma unroll
  for (int r = 0; r < 8; ++r) {
    int row = rbase + r;
    float v = c[r] * scale;
    if (bias) v += bias[bcol];
    if (act == 1)      v = eluf(v);
    else if (act == 2) v = tanhf(v);
    float* dst = Cb + (long long)row * ldc + bcol;
    if (addc) v += *dst;
    *dst = v;
  }
}

// ============================================================================
// Row LayerNorm (rows of length D), 256 threads / row
// ============================================================================
__global__ void __launch_bounds__(256)
k_layernorm(const float* __restrict__ in, const float* __restrict__ g,
            const float* __restrict__ be, float* __restrict__ out, int D)
{
  __shared__ float red[256];
  int row = blockIdx.x, tid = threadIdx.x;
  const float* r = in + (long long)row * D;
  float s = 0.f;
  for (int i = tid; i < D; i += 256) s += r[i];
  red[tid] = s; __syncthreads();
  for (int k = 128; k > 0; k >>= 1) { if (tid < k) red[tid] += red[tid + k]; __syncthreads(); }
  float mean = red[0] / D; __syncthreads();
  float v = 0.f;
  for (int i = tid; i < D; i += 256) { float d = r[i] - mean; v += d * d; }
  red[tid] = v; __syncthreads();
  for (int k = 128; k > 0; k >>= 1) { if (tid < k) red[tid] += red[tid + k]; __syncthreads(); }
  float inv = rsqrtf(red[0] / D + 1e-5f);
  for (int i = tid; i < D; i += 256)
    out[(long long)row * D + i] = (r[i] - mean) * inv * g[i] + be[i];
}

// ============================================================================
// Post-encoder: x = elu(ln(xe)) + cancer_emb ; *= 1+elu(age*w+b) ; += pos_emb
// ============================================================================
__global__ void __launch_bounds__(256)
k_postenc(const float* __restrict__ xe, const float* __restrict__ g, const float* __restrict__ be,
          const float* __restrict__ cancer_emb, const int* __restrict__ cancer_type,
          const float* __restrict__ clinical, const float* __restrict__ age_w,
          const float* __restrict__ age_b, const float* __restrict__ pos_emb,
          const int* __restrict__ cpos, float* __restrict__ x)
{
  __shared__ float red[256];
  int row = blockIdx.x, tid = threadIdx.x;
  int b = row / N_;
  const float* r = xe + (long long)row * H_;
  float s = 0.f;
  for (int i = tid; i < H_; i += 256) s += r[i];
  red[tid] = s; __syncthreads();
  for (int k = 128; k > 0; k >>= 1) { if (tid < k) red[tid] += red[tid + k]; __syncthreads(); }
  float mean = red[0] / H_; __syncthreads();
  float v = 0.f;
  for (int i = tid; i < H_; i += 256) { float d = r[i] - mean; v += d * d; }
  red[tid] = v; __syncthreads();
  for (int k = 128; k > 0; k >>= 1) { if (tid < k) red[tid] += red[tid + k]; __syncthreads(); }
  float inv = rsqrtf(red[0] / H_ + 1e-5f);
  int ct = cancer_type[b];
  float age = clinical[b * 3 + 0];
  int pi = cpos[row]; if (pi > NPOS_ - 1) pi = NPOS_ - 1;
  for (int i = tid; i < H_; i += 256) {
    float val = (r[i] - mean) * inv * g[i] + be[i];
    val = eluf(val);
    val += cancer_emb[ct * H_ + i];
    val *= (1.f + eluf(age * age_w[i] + age_b[i]));
    val += pos_emb[pi * H_ + i];
    x[(long long)row * H_ + i] = val;
  }
}

// ============================================================================
// Fused edge-MLP + mask + softmax over keys, all 8 heads at once, in-place on
// the score tensor. One block per (b, q) row; 256 threads x 2 keys each.
// Reads edge_features exactly once per layer (the dominant HBM traffic).
// ============================================================================
__global__ void __launch_bounds__(256)
k_edge_softmax(float* __restrict__ sc, const float* __restrict__ edge,
               const float* __restrict__ ew1, const float* __restrict__ eb1,
               const float* __restrict__ ew2, const float* __restrict__ eb2,
               const int* __restrict__ block_ids, const int* __restrict__ node_mask)
{
  __shared__ float w1[256], w2[128], b1[16], b2[8];
  __shared__ float red[256 * 8];
  int tid = threadIdx.x;
  if (tid < 256) w1[tid] = ew1[tid];
  if (tid < 128) w2[tid] = ew2[tid];
  if (tid < 16)  b1[tid] = eb1[tid];
  if (tid < 8)   b2[tid] = eb2[tid];
  __syncthreads();

  int b = blockIdx.x / N_;
  int q = blockIdx.x % N_;
  int  qb   = block_ids[b * N_ + q];
  bool qpad = node_mask[b * N_ + q] == 0;

  float sv[2][8];
  bool  msk[2];
  long long scq = (long long)b * NH_ * N_ * N_ + (long long)q * N_;   // + h*N*N + j

#pragma unroll
  for (int t = 0; t < 2; ++t) {
    int j = tid + t * 256;
    bool m = qpad || (node_mask[b * N_ + j] == 0) || (block_ids[b * N_ + j] != qb);
    msk[t] = m;
    float e[16];
    const float* ep = edge + (((long long)b * N_ + q) * N_ + j) * NEDGE_;
#pragma unroll
    for (int i = 0; i < 16; ++i) e[i] = ep[i];
    float eb[8];
#pragma unroll
    for (int h = 0; h < 8; ++h) eb[h] = b2[h];
#pragma unroll
    for (int tt = 0; tt < 16; ++tt) {
      float hs = b1[tt];
#pragma unroll
      for (int i = 0; i < 16; ++i) hs += e[i] * w1[i * 16 + tt];
      hs = eluf(hs);
#pragma unroll
      for (int h = 0; h < 8; ++h) eb[h] += hs * w2[tt * 8 + h];
    }
#pragma unroll
    for (int h = 0; h < 8; ++h)
      sv[t][h] = m ? NEG_INF : sc[scq + (long long)h * N_ * N_ + j] + eb[h];
  }

  // per-head max
#pragma unroll
  for (int h = 0; h < 8; ++h) red[tid * 8 + h] = fmaxf(sv[0][h], sv[1][h]);
  __syncthreads();
  for (int k = 128; k > 0; k >>= 1) {
    if (tid < k) {
#pragma unroll
      for (int h = 0; h < 8; ++h)
        red[tid * 8 + h] = fmaxf(red[tid * 8 + h], red[(tid + k) * 8 + h]);
    }
    __syncthreads();
  }
  float mh[8];
#pragma unroll
  for (int h = 0; h < 8; ++h) mh[h] = red[h];
  __syncthreads();

  // per-head sum of exp
#pragma unroll
  for (int h = 0; h < 8; ++h) {
    float s = 0.f;
    if (!msk[0]) s += __expf(sv[0][h] - mh[h]);
    if (!msk[1]) s += __expf(sv[1][h] - mh[h]);
    red[tid * 8 + h] = s;
  }
  __syncthreads();
  for (int k = 128; k > 0; k >>= 1) {
    if (tid < k) {
#pragma unroll
      for (int h = 0; h < 8; ++h) red[tid * 8 + h] += red[(tid + k) * 8 + h];
    }
    __syncthreads();
  }
  float sum[8];
#pragma unroll
  for (int h = 0; h < 8; ++h) sum[h] = red[h];

#pragma unroll
  for (int t = 0; t < 2; ++t) {
    int j = tid + t * 256;
#pragma unroll
    for (int h = 0; h < 8; ++h) {
      float p = 0.f;
      if (!msk[t] && sum[h] > 0.f) p = __expf(sv[t][h] - mh[h]) / sum[h];
      sc[scq + (long long)h * N_ * N_ + j] = p;     // nan_to_num(all -inf) -> 0
    }
  }
}

// ============================================================================
// Channel attention weights: softmax of pool scores within each channel.
// ============================================================================
__global__ void __launch_bounds__(256)
k_channel_weights(const float* __restrict__ pool_s, const int* __restrict__ channel_ids,
                  const int* __restrict__ node_mask, float* __restrict__ wgt,
                  int* __restrict__ chpad)
{
  __shared__ float red[256];
  int tid = threadIdx.x;
  int b = blockIdx.x / NCH_;
  int c = blockIdx.x % NCH_;
  float sv[2]; bool mem[2];
#pragma unroll
  for (int t = 0; t < 2; ++t) {
    int n = tid + t * 256;
    mem[t] = (channel_ids[b * N_ + n] == c) && (node_mask[b * N_ + n] != 0);
    sv[t]  = mem[t] ? pool_s[b * N_ + n] : NEG_INF;
  }
  red[tid] = fmaxf(sv[0], sv[1]); __syncthreads();
  for (int k = 128; k > 0; k >>= 1) { if (tid < k) red[tid] = fmaxf(red[tid], red[tid + k]); __syncthreads(); }
  float mx = red[0]; __syncthreads();
  float s = 0.f;
  if (mem[0]) s += __expf(sv[0] - mx);
  if (mem[1]) s += __expf(sv[1] - mx);
  red[tid] = s; __syncthreads();
  for (int k = 128; k > 0; k >>= 1) { if (tid < k) red[tid] += red[tid + k]; __syncthreads(); }
  float sm = red[0];
#pragma unroll
  for (int t = 0; t < 2; ++t) {
    int n = tid + t * 256;
    float w = (mem[t] && sm > 0.f) ? __expf(sv[t] - mx) / sm : 0.f;
    wgt[((long long)b * NCH_ + c) * N_ + n] = w;
  }
  if (tid == 0) chpad[b * NCH_ + c] = (mx == NEG_INF) ? 1 : 0;
}

// ============================================================================
// Cross-layer 16x16 masked softmax, one block per (b,h)
// ============================================================================
__global__ void __launch_bounds__(256)
k_cross_softmax(float* __restrict__ sc, const int* __restrict__ chpad)
{
  __shared__ float s[256];
  __shared__ float rmax[16], rsum[16];
  int tid = threadIdx.x;
  int b  = blockIdx.x / NH_;
  int q  = tid / 16, kk = tid % 16;
  long long base = (long long)blockIdx.x * 256;
  bool m = (chpad[b * 16 + q] != 0) || (chpad[b * 16 + kk] != 0);
  float v = m ? NEG_INF : sc[base + tid];
  s[tid] = v;
  __syncthreads();
  if (tid < 16) {
    float mx = NEG_INF;
    for (int i = 0; i < 16; ++i) mx = fmaxf(mx, s[tid * 16 + i]);
    float sm = 0.f;
    if (mx > NEG_INF)
      for (int i = 0; i < 16; ++i) sm += (s[tid * 16 + i] > NEG_INF) ? __expf(s[tid * 16 + i] - mx) : 0.f;
    rmax[tid] = mx; rsum[tid] = sm;
  }
  __syncthreads();
  float p = 0.f;
  if (!m && rsum[q] > 0.f) p = __expf(v - rmax[q]) / rsum[q];
  sc[base + tid] = p;
}

// ============================================================================
// out[row] = dot(in[row, 0:dim], w) + b0[0]
// ============================================================================
__global__ void __launch_bounds__(256)
k_dot_reduce(const float* __restrict__ in, int dim, const float* __restrict__ w,
             const float* __restrict__ b0, float* __restrict__ out)
{
  __shared__ float red[256];
  int tid = threadIdx.x;
  const float* r = in + (long long)blockIdx.x * dim;
  float s = 0.f;
  for (int i = tid; i < dim; i += 256) s += r[i] * w[i];
  red[tid] = s; __syncthreads();
  for (int k = 128; k > 0; k >>= 1) { if (tid < k) red[tid] += red[tid + k]; __syncthreads(); }
  if (tid == 0) out[blockIdx.x] = red[0] + b0[0];
}

// ============================================================================
// patient[b,:] = softmax_c(cs, mask=chpad) @ tokens[b]
// ============================================================================
__global__ void __launch_bounds__(512)
k_patient(const float* __restrict__ cs, const int* __restrict__ chpad,
          const float* __restrict__ tok, float* __restrict__ patient)
{
  int b = blockIdx.x, d = threadIdx.x;
  float mx = NEG_INF;
  for (int c = 0; c < NCH_; ++c) if (!chpad[b * NCH_ + c]) mx = fmaxf(mx, cs[b * NCH_ + c]);
  float sm = 0.f;
  if (mx > NEG_INF)
    for (int c = 0; c < NCH_; ++c) if (!chpad[b * NCH_ + c]) sm += __expf(cs[b * NCH_ + c] - mx);
  float acc = 0.f;
  for (int c = 0; c < NCH_; ++c) {
    float w = 0.f;
    if (!chpad[b * NCH_ + c] && sm > 0.f) w = __expf(cs[b * NCH_ + c] - mx) / sm;
    acc += w * tok[((long long)b * NCH_ + c) * H_ + d];
  }
  patient[(long long)b * H_ + d] = acc;
}

// ============================================================================
// Readout: feats = [patient(512) | sex*w+b (128) | elu(atlas*w+b) (128)]
//          out = elu(feats @ ro_w1 + ro_b1) @ ro_w2 + ro_b2
// ============================================================================
__global__ void __launch_bounds__(512)
k_readout(const float* __restrict__ patient, const float* __restrict__ clinical,
          const float* __restrict__ sex_w, const float* __restrict__ sex_b,
          const float* __restrict__ atlas_w, const float* __restrict__ atlas_b,
          const float* __restrict__ ro_w1, const float* __restrict__ ro_b1,
          const float* __restrict__ ro_w2, const float* __restrict__ ro_b2,
          float* __restrict__ out)
{
  __shared__ float feats[768];
  __shared__ float red[512];
  int b = blockIdx.x, tid = threadIdx.x;
  float sex   = clinical[b * 3 + 1];
  float atlas = clinical[b * 3 + 2];
  feats[tid] = patient[(long long)b * H_ + tid];
  if (tid < 128) {
    feats[512 + tid] = sex * sex_w[tid] + sex_b[tid];
    feats[640 + tid] = eluf(atlas * atlas_w[tid] + atlas_b[tid]);
  }
  __syncthreads();
  float acc = ro_b1[tid];
  for (int i = 0; i < 768; ++i) acc += feats[i] * ro_w1[i * 512 + tid];
  red[tid] = eluf(acc) * ro_w2[tid];
  __syncthreads();
  for (int k = 256; k > 0; k >>= 1) { if (tid < k) red[tid] += red[tid + k]; __syncthreads(); }
  if (tid == 0) out[b] = red[0] + ro_b2[0];
}

// ============================================================================
// Host orchestration
// ============================================================================
static inline void gemm(hipStream_t st,
                        const float* A, long long a0, long long a1, int ldaM,
                        const float* Bm, long long b0, long long b1, int ldbK, int ldbN,
                        const float* bias,
                        float* C, long long c0, long long c1, int ldc,
                        int M, int N, int K, int Z, int zdiv,
                        float scale, int act, int addc)
{
  dim3 g((N + 63) / 64, M / 16, Z);
  if ((K & 31) == 0)
    gemm_wmma<false><<<g, dim3(128), 0, st>>>(A, a0, a1, ldaM, Bm, b0, b1, ldbK, ldbN,
                                              bias, C, c0, c1, ldc, M, N, K, zdiv, scale, act, addc);
  else
    gemm_wmma<true><<<g, dim3(128), 0, st>>>(A, a0, a1, ldaM, Bm, b0, b1, ldbK, ldbN,
                                             bias, C, c0, c1, ldc, M, N, K, zdiv, scale, act, addc);
}

// params leaf order (jax tree-flatten: dict keys sorted, lists in order)
enum {
  P_AGE_B = 0, P_AGE_W, P_ATLAS_B, P_ATLAS_W, P_B2C_B, P_B2C_W,
  P_BLK0 = 6,                // 20 leaves: bk bo bq bv eb1 eb2 ew1 ew2 f1b f1w f2b f2w n1b n1g n2b n2g wk wo wq wv
  P_BLK1 = P_BLK0 + 20,
  P_CANCER = P_BLK1 + 20,    // 46
  P_CHPOOL_B1, P_CHPOOL_B2, P_CHPOOL_W1, P_CHPOOL_W2,
  P_CROSS0 = 51,             // 16 leaves: bk bo bq bv f1b f1w f2b f2w n1b n1g n2b n2g wk wo wq wv
  P_ENC_B = P_CROSS0 + 16,   // 67
  P_ENC_BETA, P_ENC_G, P_ENC_W,
  P_POOL_B1, P_POOL_B2, P_POOL_W1, P_POOL_W2,
  P_POS_EMB,                 // 75
  P_RO_B1, P_RO_B2, P_RO_W1, P_RO_W2,
  P_SEX_B, P_SEX_W           // 80, 81
};
// block-layer relative: 0 bk,1 bo,2 bq,3 bv,4 eb1,5 eb2,6 ew1,7 ew2,8 f1b,9 f1w,
//                       10 f2b,11 f2w,12 n1b,13 n1g,14 n2b,15 n2g,16 wk,17 wo,18 wq,19 wv
// cross-layer relative: 0 bk,1 bo,2 bq,3 bv,4 f1b,5 f1w,6 f2b,7 f2w,8 n1b,9 n1g,
//                       10 n2b,11 n2g,12 wk,13 wo,14 wq,15 wv

extern "C" void kernel_launch(void* const* d_in, const int* in_sizes, int n_in,
                              void* d_out, int out_size, void* d_ws, size_t ws_size,
                              hipStream_t stream)
{
  (void)in_sizes; (void)out_size; (void)ws_size;

  const float* node_features = (const float*)d_in[0];
  const int*   node_mask     = (const int*)  d_in[1];
  const int*   cpos          = (const int*)  d_in[2];
  const int*   cancer_type   = (const int*)  d_in[3];
  const float* clinical      = (const float*)d_in[4];
  const float* edge          = (const float*)d_in[5];
  const int*   block_ids     = (const int*)  d_in[6];
  const int*   channel_ids   = (const int*)  d_in[7];

  const float* P[96] = {};
  for (int i = 0; i < 96 && (8 + i) < n_in; ++i) P[i] = (const float*)d_in[8 + i];

  // ---- workspace carve ----
  char* wsb = (char*)d_ws;
  size_t off = 0;
  auto alloc = [&](size_t nfloats) -> float* {
    float* p = (float*)(wsb + off);
    off += ((nfloats * sizeof(float)) + 255) & ~(size_t)255;
    return p;
  };
  float* x    = alloc((size_t)B_ * N_ * H_);
  float* xn   = alloc((size_t)B_ * N_ * H_);
  float* qb_  = alloc((size_t)B_ * N_ * H_);
  float* kb_  = alloc((size_t)B_ * N_ * H_);
  float* vb_  = alloc((size_t)B_ * N_ * H_);
  float* ao   = alloc((size_t)B_ * N_ * H_);
  float* sc   = alloc((size_t)B_ * NH_ * N_ * N_);       // 64 MB
  float* ffh  = alloc((size_t)B_ * N_ * FF_);
  float* tp   = alloc((size_t)B_ * N_ * 256);
  float* pool_s = alloc((size_t)B_ * N_);
  float* wgt  = alloc((size_t)B_ * NCH_ * N_);
  int*   chpad = (int*)alloc((size_t)B_ * NCH_);
  float* chemb = alloc((size_t)B_ * NCH_ * H_);
  float* tok  = alloc((size_t)B_ * NCH_ * H_);
  float* tokn = alloc((size_t)B_ * NCH_ * H_);
  float* tq   = alloc((size_t)B_ * NCH_ * H_);
  float* tk   = alloc((size_t)B_ * NCH_ * H_);
  float* tv   = alloc((size_t)B_ * NCH_ * H_);
  float* tao  = alloc((size_t)B_ * NCH_ * H_);
  float* tsc  = alloc((size_t)B_ * NH_ * NCH_ * NCH_);
  float* tffh = alloc((size_t)B_ * NCH_ * FF_);
  float* csh  = alloc((size_t)B_ * NCH_ * 256);
  float* cs   = alloc((size_t)B_ * NCH_);
  float* patient = alloc((size_t)B_ * H_);

  const int M = B_ * N_;         // 4096
  const int MT = B_ * NCH_;      // 128

  // ---- encoder ----
  gemm(stream, node_features, 0, 0, 64, P[P_ENC_W], 0, 0, H_, 1, P[P_ENC_B],
       xn, 0, 0, H_, M, H_, 64, 1, 1, 1.f, 0, 0);
  k_postenc<<<B_ * N_, 256, 0, stream>>>(xn, P[P_ENC_G], P[P_ENC_BETA], P[P_CANCER],
                                         cancer_type, clinical, P[P_AGE_W], P[P_AGE_B],
                                         P[P_POS_EMB], cpos, x);

  // ---- block layers (with edge bias) ----
  auto block_layer = [&](const float* const* L) {
    k_layernorm<<<B_ * N_, 256, 0, stream>>>(x, L[13], L[12], xn, H_);
    gemm(stream, xn, 0, 0, H_, L[18], 0, 0, H_, 1, L[2], qb_, 0, 0, H_, M, H_, H_, 1, 1, 1.f, 0, 0);
    gemm(stream, xn, 0, 0, H_, L[16], 0, 0, H_, 1, L[0], kb_, 0, 0, H_, M, H_, H_, 1, 1, 1.f, 0, 0);
    gemm(stream, xn, 0, 0, H_, L[19], 0, 0, H_, 1, L[3], vb_, 0, 0, H_, M, H_, H_, 1, 1, 1.f, 0, 0);
    // s[b,h] = q[b,:,h] @ k[b,:,h]^T / 8
    gemm(stream, qb_, (long long)N_ * H_, HD_, H_,
                 kb_, (long long)N_ * H_, HD_, 1, H_,
                 nullptr, sc, (long long)NH_ * N_ * N_, (long long)N_ * N_, N_,
                 N_, N_, HD_, B_ * NH_, NH_, 0.125f, 0, 0);
    k_edge_softmax<<<B_ * N_, 256, 0, stream>>>(sc, edge, L[6], L[4], L[7], L[5],
                                                block_ids, node_mask);
    // attn @ v
    gemm(stream, sc, (long long)NH_ * N_ * N_, (long long)N_ * N_, N_,
                 vb_, (long long)N_ * H_, HD_, H_, 1,
                 nullptr, ao, (long long)N_ * H_, HD_, H_,
                 N_, HD_, N_, B_ * NH_, NH_, 1.f, 0, 0);
    // o-proj + residual
    gemm(stream, ao, 0, 0, H_, L[17], 0, 0, H_, 1, L[1], x, 0, 0, H_, M, H_, H_, 1, 1, 1.f, 0, 1);
    // FFN + residual
    k_layernorm<<<B_ * N_, 256, 0, stream>>>(x, L[15], L[14], xn, H_);
    gemm(stream, xn, 0, 0, H_, L[9], 0, 0, FF_, 1, L[8], ffh, 0, 0, FF_, M, FF_, H_, 1, 1, 1.f, 1, 0);
    gemm(stream, ffh, 0, 0, FF_, L[11], 0, 0, H_, 1, L[10], x, 0, 0, H_, M, H_, FF_, 1, 1, 1.f, 0, 1);
  };
  block_layer(&P[P_BLK0]);
  block_layer(&P[P_BLK1]);

  // ---- node pooling -> channel tokens ----
  gemm(stream, x, 0, 0, H_, P[P_POOL_W1], 0, 0, 256, 1, P[P_POOL_B1],
       tp, 0, 0, 256, M, 256, H_, 1, 1, 1.f, 2, 0);
  k_dot_reduce<<<B_ * N_, 256, 0, stream>>>(tp, 256, P[P_POOL_W2], P[P_POOL_B2], pool_s);
  k_channel_weights<<<B_ * NCH_, 256, 0, stream>>>(pool_s, channel_ids, node_mask, wgt, chpad);
  // ch_embed[b] = wgt[b] (16xN) @ x[b] (NxH)
  gemm(stream, wgt, (long long)NCH_ * N_, 0, N_,
               x, (long long)N_ * H_, 0, H_, 1,
               nullptr, chemb, (long long)NCH_ * H_, 0, H_,
               NCH_, H_, N_, B_, 1, 1.f, 0, 0);
  gemm(stream, chemb, 0, 0, H_, P[P_B2C_W], 0, 0, H_, 1, P[P_B2C_B],
       tok, 0, 0, H_, MT, H_, H_, 1, 1, 1.f, 0, 0);

  // ---- cross layer (T = 16, no edge bias) ----
  {
    const float* const* L = &P[P_CROSS0];
    k_layernorm<<<B_ * NCH_, 256, 0, stream>>>(tok, L[9], L[8], tokn, H_);
    gemm(stream, tokn, 0, 0, H_, L[14], 0, 0, H_, 1, L[2], tq, 0, 0, H_, MT, H_, H_, 1, 1, 1.f, 0, 0);
    gemm(stream, tokn, 0, 0, H_, L[12], 0, 0, H_, 1, L[0], tk, 0, 0, H_, MT, H_, H_, 1, 1, 1.f, 0, 0);
    gemm(stream, tokn, 0, 0, H_, L[15], 0, 0, H_, 1, L[3], tv, 0, 0, H_, MT, H_, H_, 1, 1, 1.f, 0, 0);
    gemm(stream, tq, (long long)NCH_ * H_, HD_, H_,
                 tk, (long long)NCH_ * H_, HD_, 1, H_,
                 nullptr, tsc, (long long)NH_ * NCH_ * NCH_, (long long)NCH_ * NCH_, NCH_,
                 NCH_, NCH_, HD_, B_ * NH_, NH_, 0.125f, 0, 0);
    k_cross_softmax<<<B_ * NH_, 256, 0, stream>>>(tsc, chpad);
    gemm(stream, tsc, (long long)NH_ * NCH_ * NCH_, (long long)NCH_ * NCH_, NCH_,
                 tv, (long long)NCH_ * H_, HD_, H_, 1,
                 nullptr, tao, (long long)NCH_ * H_, HD_, H_,
                 NCH_, HD_, NCH_, B_ * NH_, NH_, 1.f, 0, 0);
    gemm(stream, tao, 0, 0, H_, L[13], 0, 0, H_, 1, L[1], tok, 0, 0, H_, MT, H_, H_, 1, 1, 1.f, 0, 1);
    k_layernorm<<<B_ * NCH_, 256, 0, stream>>>(tok, L[11], L[10], tokn, H_);
    gemm(stream, tokn, 0, 0, H_, L[5], 0, 0, FF_, 1, L[4], tffh, 0, 0, FF_, MT, FF_, H_, 1, 1, 1.f, 1, 0);
    gemm(stream, tffh, 0, 0, FF_, L[7], 0, 0, H_, 1, L[6], tok, 0, 0, H_, MT, H_, FF_, 1, 1, 1.f, 0, 1);
  }

  // ---- channel pooling + readout ----
  gemm(stream, tok, 0, 0, H_, P[P_CHPOOL_W1], 0, 0, 256, 1, P[P_CHPOOL_B1],
       csh, 0, 0, 256, MT, 256, H_, 1, 1, 1.f, 2, 0);
  k_dot_reduce<<<B_ * NCH_, 256, 0, stream>>>(csh, 256, P[P_CHPOOL_W2], P[P_CHPOOL_B2], cs);
  k_patient<<<B_, 512, 0, stream>>>(cs, chpad, tok, patient);
  k_readout<<<B_, 512, 0, stream>>>(patient, clinical, P[P_SEX_W], P[P_SEX_B],
                                    P[P_ATLAS_W], P[P_ATLAS_B],
                                    P[P_RO_W1], P[P_RO_B1], P[P_RO_W2], P[P_RO_B2],
                                    (float*)d_out);
}
